// FloodModel_21251498180645
// MI455X (gfx1250) — matrix operand order repaired and checked by reading, not compiled, not run on hardware
//
#include <hip/hip_runtime.h>
#include <hip/hip_bf16.h>
#include <math.h>

// ---------------------------------------------------------------------------
// Model dimensions (FloodModel GVP-GNN)
// ---------------------------------------------------------------------------
static constexpr int Nn = 20000;
static constexpr int Ee = 320000;
static constexpr int Tt = 2;
static constexpr int Gg = 32;
static constexpr int S_FIXED = 16, S_DYN = 8, V_DYN = 8;
static constexpr int SI = S_FIXED + S_DYN + 1;   // 25
static constexpr int SH = 64, VH = 16;

typedef __attribute__((ext_vector_type(8)))  float   f32x8;
typedef __attribute__((ext_vector_type(8)))  __bf16  bf16x8;
typedef __attribute__((ext_vector_type(16))) __bf16  bf16x16;

__device__ __forceinline__ __bf16 f2bf(float x) {
    unsigned u = __builtin_bit_cast(unsigned, x);
    unsigned r = u + 0x7FFFu + ((u >> 16) & 1u);          // round-to-nearest-even
    unsigned short s = (unsigned short)(r >> 16);
    return __builtin_bit_cast(__bf16, s);
}

// ---------------------------------------------------------------------------
// Weight prep: W[K][Nc] (f32, row-major) -> Bt[c][k] (bf16, K-padded, col tiles)
// Rows k >= K are zero, which annihilates any garbage in the A-side padding.
// ---------------------------------------------------------------------------
__global__ void prep_weight(const float* __restrict__ W, int K, int Nc,
                            __bf16* __restrict__ Bt, int Kpad, int NcPad) {
    int i = blockIdx.x * blockDim.x + threadIdx.x;
    if (i >= NcPad * Kpad) return;
    int c = i / Kpad, k = i % Kpad;
    Bt[i] = (k < K && c < Nc) ? f2bf(W[(size_t)k * Nc + c]) : f2bf(0.0f);
}

// ---------------------------------------------------------------------------
// Core GEMM: C[M][NT*16] = relu(A[M][lda] @ B[K][NT*16] + bias)
// - bf16 WMMA 16x16x32, 8 waves/block, 16 rows/wave, NT 16-col tiles
// - B (bf16, transposed, K-padded) staged in LDS via
//   global_load_async_to_lds_b128 (ASYNCcnt) once per block
// - A fragments loaded straight from global as 4x b128 per lane per K-chunk
//   (lda >= Kpad guarantees in-bounds; zero B-rows kill the padding columns)
// - Caller guarantees Ncols == NT*16, so the epilogue has no column guards;
//   full waves take a straight-line store path.
// ---------------------------------------------------------------------------
template <int NT>
__global__ __launch_bounds__(256)
void gemm_bias_relu_wmma(const float* __restrict__ A, int lda, int K, int Kpad,
                         const __bf16* __restrict__ Bt,
                         const float* __restrict__ bias,
                         float* __restrict__ C, int ldc, int M) {
    __shared__ __align__(16) __bf16 sB[128 * 192];

    __builtin_amdgcn_s_wait_tensorcnt(0);

    const int tid = threadIdx.x;
    const int totBytes = NT * 16 * Kpad * 2;
    const unsigned ldsBase = (unsigned)(size_t)&sB[0];   // 32-bit LDS offset
    for (int i = tid * 16; i < totBytes; i += 256 * 16) {
        unsigned lds = ldsBase + (unsigned)i;
        unsigned long long ga = (unsigned long long)(const void*)Bt + (unsigned)i;
        asm volatile("global_load_async_to_lds_b128 %0, %1, off"
                     :: "v"(lds), "v"(ga) : "memory");
    }
    asm volatile("s_wait_asynccnt 0" ::: "memory");
    __syncthreads();

    const int wave = tid >> 5;
    const int lane = tid & 31;
    const int half = lane >> 4;       // 0: K-low half, 1: K-high half
    const int r16  = lane & 15;
    const int rowBase = blockIdx.x * 128 + wave * 16;
    const int row  = rowBase + r16;
    const int lrow = (row < M) ? row : (M - 1);          // clamp: loads stay legal
    const float* __restrict__ arow = A + (size_t)lrow * lda;

    f32x8 acc[NT];
#pragma unroll
    for (int t = 0; t < NT; ++t)
        acc[t] = f32x8{0.f, 0.f, 0.f, 0.f, 0.f, 0.f, 0.f, 0.f};

    for (int k0 = 0; k0 < K; k0 += 32) {
        // ---- A fragment: row=r16, K = 16h..16h+15 (contiguous) --------
        const float* ap = arow + k0 + 16 * half;
        float fa[16];
        *(float4*)(fa + 0)  = *(const float4*)(ap + 0);
        *(float4*)(fa + 4)  = *(const float4*)(ap + 4);
        *(float4*)(fa + 8)  = *(const float4*)(ap + 8);
        *(float4*)(fa + 12) = *(const float4*)(ap + 12);
        bf16x16 afrag;
#pragma unroll
        for (int j = 0; j < 16; ++j) afrag[j] = f2bf(fa[j]);

#pragma unroll
        for (int t = 0; t < NT; ++t) {
            // B fragment: col = t*16+r16, K = 16*half .. +15 (contiguous in sB)
            const __bf16* bp = &sB[(t * 16 + r16) * Kpad + k0 + 16 * half];
            bf16x8 blo = *(const bf16x8*)bp;
            bf16x8 bhi = *(const bf16x8*)(bp + 8);
            bf16x16 bfrag = __builtin_shufflevector(blo, bhi,
                0, 1, 2, 3, 4, 5, 6, 7, 8, 9, 10, 11, 12, 13, 14, 15);
            acc[t] = __builtin_amdgcn_wmma_f32_16x16x32_bf16(
                false, afrag, false, bfrag, (short)0, acc[t], false, false);
        }
    }

    // ---- epilogue: D layout -> lanes 0-15 M=i, lanes 16-31 M=8+i ----------
    if (rowBase + 16 <= M) {                 // wave fully in range: no guards
#pragma unroll
        for (int t = 0; t < NT; ++t) {
            const float b = bias[t * 16 + r16];
#pragma unroll
            for (int i = 0; i < 8; ++i) {
                const int orow = rowBase + half * 8 + i;
                C[(size_t)orow * ldc + t * 16 + r16] = fmaxf(acc[t][i] + b, 0.0f);
            }
        }
    } else {
#pragma unroll
        for (int t = 0; t < NT; ++t) {
            const float b = bias[t * 16 + r16];
#pragma unroll
            for (int i = 0; i < 8; ++i) {
                const int orow = rowBase + half * 8 + i;
                if (orow < M)
                    C[(size_t)orow * ldc + t * 16 + r16] = fmaxf(acc[t][i] + b, 0.0f);
            }
        }
    }
}

// ---------------------------------------------------------------------------
// Vector channel path: vh[m][h][c] = sum_v V[m][v][c] * Wh[v][h]; vn = ||vh||
// ---------------------------------------------------------------------------
__global__ void vec_hidden(const float* __restrict__ V, const float* __restrict__ Wh,
                           float* __restrict__ VHout, float* __restrict__ VN,
                           long M, int vi, int h) {
    long idx = (long)blockIdx.x * blockDim.x + threadIdx.x;
    if (idx >= M * h) return;
    long m = idx / h; int j = (int)(idx % h);
    float a0 = 0.f, a1 = 0.f, a2 = 0.f;
    const float* vp = V + (size_t)m * vi * 3;
    for (int r = 0; r < vi; ++r) {
        float w = Wh[r * h + j];
        a0 += vp[r * 3 + 0] * w; a1 += vp[r * 3 + 1] * w; a2 += vp[r * 3 + 2] * w;
    }
    size_t o = (size_t)m * h * 3 + (size_t)j * 3;
    VHout[o] = a0; VHout[o + 1] = a1; VHout[o + 2] = a2;
    VN[(size_t)m * h + j] = sqrtf(fmaxf(a0 * a0 + a1 * a1 + a2 * a2, 1e-8f));
}

// v_out = vh @ Wv; mode 1 = sigmoid(||.||) gate; mode 2 = normalize (+ norm out)
__global__ void vec_out(const float* __restrict__ VHin, const float* __restrict__ Wv,
                        float* __restrict__ Vout, float* __restrict__ VNout,
                        long M, int h, int vo, int mode) {
    long idx = (long)blockIdx.x * blockDim.x + threadIdx.x;
    if (idx >= M * vo) return;
    long m = idx / vo; int o = (int)(idx % vo);
    float a0 = 0.f, a1 = 0.f, a2 = 0.f;
    const float* vp = VHin + (size_t)m * h * 3;
    for (int r = 0; r < h; ++r) {
        float w = Wv[r * vo + o];
        a0 += vp[r * 3 + 0] * w; a1 += vp[r * 3 + 1] * w; a2 += vp[r * 3 + 2] * w;
    }
    float nrm = sqrtf(fmaxf(a0 * a0 + a1 * a1 + a2 * a2, 1e-8f));
    if (mode == 1) {
        float g = 1.f / (1.f + expf(-nrm));
        a0 *= g; a1 *= g; a2 *= g;
    } else if (mode == 2) {
        float inv = 1.f / nrm;
        a0 *= inv; a1 *= inv; a2 *= inv;
        if (VNout) VNout[idx] = nrm;
    }
    size_t o3 = (size_t)m * vo * 3 + (size_t)o * 3;
    Vout[o3] = a0; Vout[o3 + 1] = a1; Vout[o3 + 2] = a2;
}

// ---------------------------------------------------------------------------
// Data movement helpers (scalar + float4-vectorized variants)
// ---------------------------------------------------------------------------
__global__ void copy_cols(const float* __restrict__ src, int sld, int soff,
                          float* __restrict__ dst, int dld, int doff,
                          long M, int d) {
    long i = (long)blockIdx.x * blockDim.x + threadIdx.x;
    if (i >= M * d) return;
    long m = i / d; int j = (int)(i % d);
    dst[(size_t)m * dld + doff + j] = src[(size_t)m * sld + soff + j];
}

__global__ void copy_cols4(const float4* __restrict__ src, int sld4, int soff4,
                           float4* __restrict__ dst, int dld4, int doff4,
                           long M, int d4) {
    long i = (long)blockIdx.x * blockDim.x + threadIdx.x;
    if (i >= M * d4) return;
    long m = i / d4; int j = (int)(i % d4);
    dst[(size_t)m * dld4 + doff4 + j] = src[(size_t)m * sld4 + soff4 + j];
}

__global__ void gather_rows4(const float4* __restrict__ src, int sld4,
                             const int* __restrict__ idx,
                             float4* __restrict__ dst, int dld4, int doff4,
                             long M, int d4) {
    long i = (long)blockIdx.x * blockDim.x + threadIdx.x;
    if (i >= M * d4) return;
    long e = i / d4; int j = (int)(i % d4);
    dst[(size_t)e * dld4 + doff4 + j] = src[(size_t)idx[e] * sld4 + j];
}

__global__ void scatter_add_rows(const float* __restrict__ msg, int sld,
                                 const int* __restrict__ idx,
                                 float* __restrict__ dst, int dld,
                                 long M, int d) {
    long i = (long)blockIdx.x * blockDim.x + threadIdx.x;
    if (i >= M * d) return;
    long e = i / d; int j = (int)(i % d);
    atomicAdd(&dst[(size_t)idx[e] * dld + j], msg[(size_t)e * sld + j]);
}

__global__ void zero_f32(float* __restrict__ p, long n) {
    long i = (long)blockIdx.x * blockDim.x + threadIdx.x;
    if (i < n) p[i] = 0.0f;
}

__global__ void gemv_add(const float* __restrict__ X, int ld,
                         const float* __restrict__ w, const float* __restrict__ b,
                         float* __restrict__ y, long M, int K) {
    long m = (long)blockIdx.x * blockDim.x + threadIdx.x;
    if (m >= M) return;
    float acc = b[0];
    const float* xp = X + (size_t)m * ld;
    for (int k = 0; k < K; ++k) acc += xp[k] * w[k];
    y[m] = acc;
}

// loss accumulation, wave32 shuffle reduction before the atomic
__global__ void loss_accum(const float* __restrict__ P, int pld,
                           const float* __restrict__ Tg, int tld, int toff,
                           float* __restrict__ loss, long M, int d,
                           float scale, int mode) {
    long i = (long)blockIdx.x * blockDim.x + threadIdx.x;
    float c = 0.0f;
    if (i < M * d) {
        long m = i / d; int j = (int)(i % d);
        float p = P[(size_t)m * pld + j];
        float t = Tg[(size_t)m * tld + toff + j];
        c = (mode == 0) ? fabsf(p - t) : expf(p - 2.0f) * fabsf(p - t);
        c *= scale;
    }
    for (int s = 16; s > 0; s >>= 1) c += __shfl_down(c, s, 32);
    if ((threadIdx.x & 31) == 0 && c != 0.0f) atomicAdd(loss, c);
}

// ---------------------------------------------------------------------------
// Host-side orchestration
// ---------------------------------------------------------------------------
static inline dim3 gsz(long n) { return dim3((unsigned)((n + 255) / 256)); }

static void launch_gemm(hipStream_t s, const float* A, int lda, int K, int Kpad,
                        const __bf16* Bt, const float* bias, float* C, int ldc,
                        long M, int nt) {
    dim3 g((unsigned)((M + 127) / 128)), b(256);
    if (nt == 8)
        gemm_bias_relu_wmma<8><<<g, b, 0, s>>>(A, lda, K, Kpad, Bt, bias, C, ldc, (int)M);
    else
        gemm_bias_relu_wmma<4><<<g, b, 0, s>>>(A, lda, K, Kpad, Bt, bias, C, ldc, (int)M);
}

static void copyc(hipStream_t st, const float* s, int sld, int soff,
                  float* d, int dld, int doff, long M, int dd) {
    if ((dd & 3) == 0 && (soff & 3) == 0 && (doff & 3) == 0 &&
        (sld & 3) == 0 && (dld & 3) == 0)
        copy_cols4<<<gsz(M * (dd / 4)), 256, 0, st>>>(
            (const float4*)s, sld / 4, soff / 4, (float4*)d, dld / 4, doff / 4, M, dd / 4);
    else
        copy_cols<<<gsz(M * dd), 256, 0, st>>>(s, sld, soff, d, dld, doff, M, dd);
}

static void gatherc(hipStream_t st, const float* s, int sld, const int* idx,
                    float* d, int dld, int doff, long M, int dd) {
    // all call sites have sld/dld/doff/dd % 4 == 0 and 16B-aligned bases
    gather_rows4<<<gsz(M * (dd / 4)), 256, 0, st>>>(
        (const float4*)s, sld / 4, idx, (float4*)d, dld / 4, doff / 4, M, dd / 4);
}

extern "C" void kernel_launch(void* const* d_in, const int* in_sizes, int n_in,
                              void* d_out, int out_size, void* d_ws, size_t ws_size,
                              hipStream_t stream) {
    (void)in_sizes; (void)n_in; (void)out_size; (void)ws_size;
    const int*   edge_index = (const int*)d_in[0];          // [2][E]
    const float* s_fixed    = (const float*)d_in[1];        // [N][16]
    const float* s_dyn      = (const float*)d_in[2];        // [N][T+1][8]
    const float* v_dyn      = (const float*)d_in[3];        // [N][T+1][8][3]
    const float* wdfp       = (const float*)d_in[4];        // [N][T+1][1]
    const int*   s_batch    = (const int*)d_in[5];
    (void)s_batch;

    // jax pytree leaf order (sorted dict keys):
    // feat_pred{wh,ws_b,ws_w,wv}, label_pred{wh,ws_b,ws_w},
    // layers[0..1]{m_gvp{..}, n_encode{..}, u_gvp{..}}, ln_b, ln_w
    int pi = 6;
    auto F = [&](int i) { return (const float*)d_in[i]; };
    const float* fp_wh  = F(pi++); const float* fp_wsb = F(pi++);
    const float* fp_wsw = F(pi++); const float* fp_wv  = F(pi++);
    (void)fp_wsb; (void)fp_wsw;
    const float* lp_wh  = F(pi++); const float* lp_wsb = F(pi++);
    const float* lp_wsw = F(pi++);
    struct GvpW { const float *wh, *wsb, *wsw, *wv; };
    struct LayerW { GvpW m, ne, u; };
    LayerW L[2];
    for (int l = 0; l < 2; ++l) {
        L[l].m  = {F(pi), F(pi + 1), F(pi + 2), F(pi + 3)}; pi += 4;
        L[l].ne = {F(pi), F(pi + 1), F(pi + 2), F(pi + 3)}; pi += 4;
        L[l].u  = {F(pi), F(pi + 1), F(pi + 2), F(pi + 3)}; pi += 4;
    }
    const float* ln_b = F(pi++);
    const float* ln_w = F(pi++);

    // ---- workspace bump allocator ----
    char* ws = (char*)d_ws;
    size_t off = 0;
    auto alloc = [&](size_t bytes) -> void* {
        off = (off + 255) & ~(size_t)255;
        void* p = ws + off; off += bytes; return p;
    };
    auto allocf = [&](size_t n) { return (float*)alloc(n * sizeof(float)); };
    auto allocb = [&](size_t n) { return (__bf16*)alloc(n * sizeof(__bf16)); };

    // bf16 transposed/padded weights for all WMMA GEMMs
    __bf16* W_ne[2];  int K_ne[2] = {SI + 16, SH + 16}, Kp_ne[2] = {64, 96};
    __bf16* W_m[2];   __bf16* W_u[2];
    for (int l = 0; l < 2; ++l) {
        W_ne[l] = allocb((size_t)64 * Kp_ne[l]);
        W_m[l]  = allocb((size_t)64 * 160);
        W_u[l]  = allocb((size_t)64 * 160);
        prep_weight<<<gsz(64 * Kp_ne[l]), 256, 0, stream>>>(L[l].ne.wsw, K_ne[l], 64, W_ne[l], Kp_ne[l], 64);
        prep_weight<<<gsz(64 * 160), 256, 0, stream>>>(L[l].m.wsw, 160, 64, W_m[l], 160, 64);
        prep_weight<<<gsz(64 * 160), 256, 0, stream>>>(L[l].u.wsw, 160, 64, W_u[l], 160, 64);
    }
    __bf16* W_lp = allocb((size_t)128 * 96);
    prep_weight<<<gsz(128 * 96), 256, 0, stream>>>(lp_wsw, 80, 128, W_lp, 96, 128);

    // node state buffers (GEMM A-inputs allocated with lda == Kpad)
    float* s_h    = allocf((size_t)Nn * 8);
    float* v_h    = allocf((size_t)Nn * 24);
    float* wdfp_h = allocf((size_t)Nn);
    float* ss     = allocf((size_t)Nn * 64);     // node scalar features (<=64)
    float* vv     = allocf((size_t)Nn * 48);     // node vector features (<=16x3)
    float* s_in   = allocf((size_t)Nn * 96);     // n_encode GEMM input, lda=96
    float* nvh    = allocf((size_t)Nn * 48);     // n_encode hidden vectors 16x3
    float* nvn    = allocf((size_t)Nn * 16);
    float* ns1    = allocf((size_t)Nn * 64);
    float* nv1    = allocf((size_t)Nn * 48);
    float* s_agg  = allocf((size_t)Nn * 64);
    float* v_agg  = allocf((size_t)Nn * 48);
    float* us2    = allocf((size_t)Nn * 160);    // [s1 | s_agg | uvn], lda=160
    float* uv     = allocf((size_t)Nn * 96);     // [v1 | v_agg]
    float* uvh    = allocf((size_t)Nn * 96);     // 32x3
    float* uvn    = allocf((size_t)Nn * 32);
    float* pvh    = allocf((size_t)Nn * 48);     // pred hidden 16x3
    float* pvn    = allocf((size_t)Nn * 16);
    float* ps_in  = allocf((size_t)Nn * 96);     // pred GEMM input, lda=96
    float* sl     = allocf((size_t)Nn * 128);
    float* fvo    = allocf((size_t)Nn * 24);     // feat_pred v_out 8x3
    // edge buffers
    float* es2 = allocf((size_t)Ee * 160);       // [s1[dst] | s1[src] | evn], lda=160
    float* ev  = allocf((size_t)Ee * 96);        // [v1[dst] | v1[src]] 32x3
    float* evh = allocf((size_t)Ee * 96);        // hidden 32x3
    float* evn = allocf((size_t)Ee * 32);
    float* sm  = allocf((size_t)Ee * 64);
    float* vm  = ev;                             // reuse ev for message vectors

    const int* e_src = edge_index;
    const int* e_dst = edge_index + Ee;

    float* out       = (float*)d_out;
    float* out_lab   = out;
    float* out_sf    = out + (size_t)Nn * Tt;
    float* out_vf    = out_sf + (size_t)Nn * Tt * 8;
    float* out_loss  = out_vf + (size_t)Nn * Tt * 24;
    zero_f32<<<1, 32, 0, stream>>>(out_loss, 1);

    // initial state: s_h = s[:,0], v_h = v[:,0], wdfp_h = wdfp[:,0]
    copyc(stream, s_dyn, (Tt + 1) * 8, 0, s_h, 8, 0, Nn, 8);
    copyc(stream, v_dyn, (Tt + 1) * 24, 0, v_h, 24, 0, Nn, 24);
    copyc(stream, wdfp, (Tt + 1), 0, wdfp_h, 1, 0, Nn, 1);

    for (int t = 1; t <= Tt; ++t) {
        // ss = concat(s_fixed, s_h, wdfp_h) [N][25]; vv = v_h [N][8][3]
        copyc(stream, s_fixed, 16, 0, ss, 64, 0, Nn, 16);
        copyc(stream, s_h, 8, 0, ss, 64, 16, Nn, 8);
        copyc(stream, wdfp_h, 1, 0, ss, 64, 24, Nn, 1);
        copyc(stream, v_h, 24, 0, vv, 24, 0, Nn, 24);

        int cur_si = SI, cur_vi = V_DYN;   // 25, 8
        for (int l = 0; l < 2; ++l) {
            const LayerW& P = L[l];
            // ---- n_encode ----
            vec_hidden<<<gsz((long)Nn * 16), 256, 0, stream>>>(vv, P.ne.wh, nvh, nvn, Nn, cur_vi, 16);
            copyc(stream, ss, 64, 0, s_in, 96, 0, Nn, cur_si);
            copyc(stream, nvn, 16, 0, s_in, 96, cur_si, Nn, 16);
            launch_gemm(stream, s_in, 96, K_ne[l], Kp_ne[l], W_ne[l], P.ne.wsb, ns1, 64, Nn, 4);
            vec_out<<<gsz((long)Nn * 16), 256, 0, stream>>>(nvh, P.ne.wv, nv1, nullptr, Nn, 16, 16, 1);

            // ---- m_gvp (edges) ----
            gatherc(stream, ns1, 64, e_dst, es2, 160, 0, Ee, 64);
            gatherc(stream, ns1, 64, e_src, es2, 160, 64, Ee, 64);
            gatherc(stream, nv1, 48, e_dst, ev, 96, 0, Ee, 48);
            gatherc(stream, nv1, 48, e_src, ev, 96, 48, Ee, 48);
            vec_hidden<<<gsz((long)Ee * 32), 256, 0, stream>>>(ev, P.m.wh, evh, evn, Ee, 32, 32);
            copyc(stream, evn, 32, 0, es2, 160, 128, Ee, 32);
            launch_gemm(stream, es2, 160, 160, 160, W_m[l], P.m.wsb, sm, 64, Ee, 4);
            vec_out<<<gsz((long)Ee * 16), 256, 0, stream>>>(evh, P.m.wv, vm, nullptr, Ee, 32, 16, 1);

            // ---- aggregate (segment_sum over dst) ----
            zero_f32<<<gsz((long)Nn * 64), 256, 0, stream>>>(s_agg, (long)Nn * 64);
            zero_f32<<<gsz((long)Nn * 48), 256, 0, stream>>>(v_agg, (long)Nn * 48);
            scatter_add_rows<<<gsz((long)Ee * 64), 256, 0, stream>>>(sm, 64, e_dst, s_agg, 64, Ee, 64);
            scatter_add_rows<<<gsz((long)Ee * 48), 256, 0, stream>>>(vm, 48, e_dst, v_agg, 48, Ee, 48);

            // ---- u_gvp ----
            copyc(stream, nv1, 48, 0, uv, 96, 0, Nn, 48);
            copyc(stream, v_agg, 48, 0, uv, 96, 48, Nn, 48);
            vec_hidden<<<gsz((long)Nn * 32), 256, 0, stream>>>(uv, P.u.wh, uvh, uvn, Nn, 32, 32);
            copyc(stream, ns1, 64, 0, us2, 160, 0, Nn, 64);
            copyc(stream, s_agg, 64, 0, us2, 160, 64, Nn, 64);
            copyc(stream, uvn, 32, 0, us2, 160, 128, Nn, 32);
            launch_gemm(stream, us2, 160, 160, 160, W_u[l], P.u.wsb, ss, 64, Nn, 4);
            vec_out<<<gsz((long)Nn * 16), 256, 0, stream>>>(uvh, P.u.wv, vv, nullptr, Nn, 32, 16, 1);

            cur_si = SH; cur_vi = VH;
        }

        // ---- label_pred: relu GEMM (80 -> 128), then linear head ----
        vec_hidden<<<gsz((long)Nn * 16), 256, 0, stream>>>(vv, lp_wh, pvh, pvn, Nn, 16, 16);
        copyc(stream, ss, 64, 0, ps_in, 96, 0, Nn, 64);
        copyc(stream, pvn, 16, 0, ps_in, 96, 64, Nn, 16);
        launch_gemm(stream, ps_in, 96, 80, 96, W_lp, lp_wsb, sl, 128, Nn, 8);
        gemv_add<<<gsz((long)Nn), 256, 0, stream>>>(sl, 128, ln_w, ln_b, wdfp_h, Nn, 128);

        // ---- feat_pred: only the vector branch is consumed ----
        vec_hidden<<<gsz((long)Nn * 16), 256, 0, stream>>>(vv, fp_wh, pvh, pvn, Nn, 16, 16);
        vec_out<<<gsz((long)Nn * 8), 256, 0, stream>>>(pvh, fp_wv, fvo, s_h, Nn, 16, 8, 2);
        copyc(stream, fvo, 24, 0, v_h, 24, 0, Nn, 24);

        // ---- outputs for this timestep ----
        copyc(stream, wdfp_h, 1, 0, out_lab, Tt, t - 1, Nn, 1);
        copyc(stream, s_h, 8, 0, out_sf, Tt * 8, (t - 1) * 8, Nn, 8);
        copyc(stream, v_h, 24, 0, out_vf, Tt * 24, (t - 1) * 24, Nn, 24);

        // ---- losses: (sum over nodes)/G, averaged over T; feat halves /2 ----
        const float sl_lab  = 1.0f / (Gg * Tt);
        const float sl_feat = 1.0f / (2.0f * Gg * Tt);
        loss_accum<<<gsz((long)Nn), 256, 0, stream>>>(wdfp_h, 1, wdfp, Tt + 1, t, out_loss, Nn, 1, sl_lab, 0);
        loss_accum<<<gsz((long)Nn * 8), 256, 0, stream>>>(s_h, 8, s_dyn, (Tt + 1) * 8, t * 8, out_loss, Nn, 8, sl_feat, 1);
        loss_accum<<<gsz((long)Nn * 24), 256, 0, stream>>>(v_h, 24, v_dyn, (Tt + 1) * 24, t * 24, out_loss, Nn, 24, sl_feat, 1);
    }
}
